// MSDeformAttn_39857296507650
// MI455X (gfx1250) — matrix operand order repaired and checked by reading, not compiled, not run on hardware
//
#include <hip/hip_runtime.h>
#include <hip/hip_bf16.h>

// ---------------- problem constants (from reference) ----------------
#define N_BATCH   2
#define LEN_Q     13294
#define D_MODEL   256
#define N_HEADS   8
#define N_LEVELS  4
#define N_POINTS  4
#define D_HEAD    32                       // == wave32 lanes
#define M_ROWS    (N_BATCH * LEN_Q)        // 26588 GEMM rows
#define NC_OFF    (N_HEADS * N_LEVELS * N_POINTS * 2)   // 256
#define NC_ATTN   (N_HEADS * N_LEVELS * N_POINTS)       // 128
#define M_TILES   ((M_ROWS + 15) / 16)     // 1662
#define N_TILES   (NC_OFF / 16 + NC_ATTN / 16)          // 16 + 8 = 24
#define NT_GROUP  6                        // N-tiles per block (24 = 4 groups of 6)
#define WAVES_PB  8                        // waves (M-tiles) per block
#define KSTEPS    8                        // K = 256 = 8 x 32

typedef __attribute__((ext_vector_type(16))) _Float16 v16h;
typedef __attribute__((ext_vector_type(8)))  float    v8f;

// ---------------------------------------------------------------------------
// Kernel 0: one-time weight pack.  fp32 [W_off | W_attn] -> f16 in this file's (contiguous-k)
// WMMA B-operand lane layout:  packed[tileN][kstep][lane] = 16 f16 (32 B),
// where col = col0 + lane%16 and K = kstep*32 + (lane>=16 ? 16 : 0) + j.
// NCOLS is a template constant so the 16 strided loads get immediate offsets.
// ---------------------------------------------------------------------------
template <int NCOLS>
__device__ inline v16h load_bcol(const float* __restrict__ W, int kbB, int ncol) {
    const float* p = W + (size_t)kbB * NCOLS + ncol;
    v16h B;
    #pragma unroll
    for (int j = 0; j < 16; ++j)
        B[j] = (_Float16)p[(size_t)j * NCOLS];
    return B;
}

__global__ __launch_bounds__(256) void pack_b(
    const float* __restrict__ Woff, const float* __restrict__ Wattn,
    _Float16* __restrict__ packed)          // [N_TILES][8][32][16] f16
{
    const int tile = blockIdx.x;            // 0 .. 23
    const int s    = threadIdx.x >> 5;      // kstep 0..7
    const int lane = threadIdx.x & 31;
    const int half = lane >> 4;
    const int l16  = lane & 15;
    const int kbB  = s * 32 + (half ? 16 : 0);

    v16h B;
    if (tile < NC_OFF / 16) {
        B = load_bcol<NC_OFF >(Woff,  kbB, tile * 16 + l16);
    } else {
        B = load_bcol<NC_ATTN>(Wattn, kbB, (tile - NC_OFF / 16) * 16 + l16);
    }
    *(v16h*)(packed + (size_t)(((tile * 8 + s) * 32) + lane) * 16) = B;
}

// ---------------------------------------------------------------------------
// Kernel 1: projection GEMM.  8 waves/block = 8 M-tiles; block shares one
// N-group of 6 tiles.  Double-buffered LDS staging of packed B (one barrier
// per K-step); per step each wave loads A once, prefetches all 6 B tiles into
// registers, then issues 6 back-to-back accumulating WMMAs reusing A.
// ---------------------------------------------------------------------------
__global__ __launch_bounds__(256) void proj_wmma(
    const float* __restrict__ q,
    const _Float16* __restrict__ packedB,
    const float* __restrict__ boff, const float* __restrict__ battn,
    float* __restrict__ off_out, float* __restrict__ attn_out)
{
    __shared__ v16h smemB[2][NT_GROUP * 32];   // 2 x 6 KB double buffer

    const int wv   = threadIdx.x >> 5;      // wave in block -> M-tile
    const int lane = threadIdx.x & 31;
    const int half = lane >> 4;
    const int l16  = lane & 15;
    const int ng   = blockIdx.y;            // N-group 0..3

    int tileM = blockIdx.x * WAVES_PB + wv; // clamped dup-waves write identical
    if (tileM > M_TILES - 1) tileM = M_TILES - 1;   // data -> barrier-safe

    int mrow = tileM * 16 + l16;
    if (mrow > M_ROWS - 1) mrow = M_ROWS - 1;
    const float* arow = q + (size_t)mrow * D_MODEL;

    // staging role for this thread (first 192 threads copy one 32 B chunk)
    const bool stager  = threadIdx.x < NT_GROUP * 32;
    const int  st_nt   = threadIdx.x >> 5;
    const int  st_ln   = threadIdx.x & 31;
    const _Float16* st_src =
        packedB + (size_t)(((ng * NT_GROUP + st_nt) * KSTEPS) * 32 + st_ln) * 16;

    v8f acc[NT_GROUP] = {};

    // prologue: stage kstep 0 into buffer 0
    if (stager)
        smemB[0][st_nt * 32 + st_ln] = *(const v16h*)(st_src);

    for (int s = 0; s < KSTEPS; ++s) {
        __syncthreads();                    // buf[s&1] ready; prev reads done

        // stage next K-step into the other buffer (overlaps this step's WMMAs)
        if (s + 1 < KSTEPS && stager)
            smemB[(s + 1) & 1][st_nt * 32 + st_ln] =
                *(const v16h*)(st_src + (size_t)(s + 1) * 32 * 16);

        // ---- A tile (16x32 f16): half 0 holds K {kb..kb+15},
        //      half 1 holds K {kb+16..kb+31}; row = lane%16.
        const int kb = s * 32;
        const int a0 = kb + (half ? 16 : 0);
        const int a1 = a0 + 8;
        const float4 f0 = *(const float4*)(arow + a0);
        const float4 f1 = *(const float4*)(arow + a0 + 4);
        const float4 f2 = *(const float4*)(arow + a1);
        const float4 f3 = *(const float4*)(arow + a1 + 4);
        v16h A;
        A[0]  = (_Float16)f0.x; A[1]  = (_Float16)f0.y; A[2]  = (_Float16)f0.z; A[3]  = (_Float16)f0.w;
        A[4]  = (_Float16)f1.x; A[5]  = (_Float16)f1.y; A[6]  = (_Float16)f1.z; A[7]  = (_Float16)f1.w;
        A[8]  = (_Float16)f2.x; A[9]  = (_Float16)f2.y; A[10] = (_Float16)f2.z; A[11] = (_Float16)f2.w;
        A[12] = (_Float16)f3.x; A[13] = (_Float16)f3.y; A[14] = (_Float16)f3.z; A[15] = (_Float16)f3.w;

        // ---- prefetch all 6 B tiles into registers (waits coalesce), then
        //      issue 6 accumulating WMMAs back-to-back reusing the same A.
        v16h Breg[NT_GROUP];
        #pragma unroll
        for (int nt = 0; nt < NT_GROUP; ++nt)
            Breg[nt] = smemB[s & 1][nt * 32 + lane];

        #pragma unroll
        for (int nt = 0; nt < NT_GROUP; ++nt)
            acc[nt] = __builtin_amdgcn_wmma_f32_16x16x32_f16(
                /*neg_a=*/false, A, /*neg_b=*/false, Breg[nt],
                /*c_mod=*/(short)0, acc[nt], /*reuse_a=*/false, /*reuse_b=*/false);
    }

    // Epilogue: C layout -> row = vgpr + 8*half, col = lane%16.  Bias + store.
    #pragma unroll
    for (int nt = 0; nt < NT_GROUP; ++nt) {
        const int tileN = ng * NT_GROUP + nt;
        const float* bias; float* dst; int ncols, col0;
        if (tileN < NC_OFF / 16) {
            bias = boff;  dst = off_out;  ncols = NC_OFF;  col0 = tileN * 16;
        } else {
            bias = battn; dst = attn_out; ncols = NC_ATTN; col0 = (tileN - NC_OFF / 16) * 16;
        }
        const int   ncol = col0 + l16;
        const float bv   = bias[ncol];
        #pragma unroll
        for (int r = 0; r < 8; ++r) {
            const int m = tileM * 16 + r + (half ? 8 : 0);
            if (m < M_ROWS)
                dst[(size_t)m * ncols + ncol] = acc[nt][r] + bv;
        }
    }
}

// ---------------------------------------------------------------------------
// Kernel 2: softmax + multi-scale bilinear sampling.
// One wave per (batch, query, head); lane == channel (D_HEAD == 32).
// All address math is wave-uniform; each corner gather is a coalesced 128B read.
// ---------------------------------------------------------------------------
__global__ __launch_bounds__(256) void msda_sample(
    const float* __restrict__ value,        // (2, 13294, 256)
    const float* __restrict__ ref,          // (2, 13294, 4, 2)
    const float* __restrict__ off,          // ws: (26588, 256)
    const float* __restrict__ attn_logits,  // ws: (26588, 128)
    float* __restrict__ out)                // (2, 13294, 256)
{
    const int wid  = blockIdx.x * (blockDim.x >> 5) + (threadIdx.x >> 5);
    const int lane = threadIdx.x & 31;

    const int h = wid & (N_HEADS - 1);
    const int q = (wid >> 3) % LEN_Q;
    const int b = wid / (N_HEADS * LEN_Q);
    if (b >= N_BATCH) return;

    const int Hs[N_LEVELS]     = {100, 50, 25, 13};
    const int Ws[N_LEVELS]     = {100, 50, 25, 13};
    const int starts[N_LEVELS] = {0, 10000, 12500, 13125};

    const size_t bq = (size_t)b * LEN_Q + q;

    // ---- softmax over the 16 (level,point) logits of this head (wave-uniform)
    const float* lg = attn_logits + bq * NC_ATTN + h * 16;
    float l[16];
    #pragma unroll
    for (int j = 0; j < 16; j += 4) {
        const float4 t = *(const float4*)(lg + j);
        l[j] = t.x; l[j + 1] = t.y; l[j + 2] = t.z; l[j + 3] = t.w;
    }
    float mx = l[0];
    #pragma unroll
    for (int j = 1; j < 16; ++j) mx = fmaxf(mx, l[j]);
    float sum = 0.f;
    #pragma unroll
    for (int j = 0; j < 16; ++j) { l[j] = __expf(l[j] - mx); sum += l[j]; }
    const float inv = 1.f / sum;

    // ---- sampling offsets: 32 contiguous floats for this (b,q,h)
    const float* ob = off + bq * NC_OFF + h * (N_LEVELS * N_POINTS * 2);
    float ox[16], oy[16];
    #pragma unroll
    for (int j = 0; j < 16; j += 2) {
        const float4 t = *(const float4*)(ob + 2 * j);
        ox[j]     = t.x; oy[j]     = t.y;
        ox[j + 1] = t.z; oy[j + 1] = t.w;
    }

    float acc = 0.f;
    #pragma unroll
    for (int lvl = 0; lvl < N_LEVELS; ++lvl) {
        const int   H = Hs[lvl], W = Ws[lvl];
        const float rx = ref[bq * (N_LEVELS * 2) + lvl * 2 + 0];
        const float ry = ref[bq * (N_LEVELS * 2) + lvl * 2 + 1];
        // channel-resolved base pointer for this level / head / lane
        const float* vbase = value + ((size_t)b * LEN_Q + starts[lvl]) * D_MODEL
                                   + h * D_HEAD + lane;
        #pragma unroll
        for (int p = 0; p < N_POINTS; ++p) {
            const int   idx = lvl * N_POINTS + p;
            // loc in [0,1]; grid = 2*loc-1; pixel x = ((gx+1)*W-1)/2 = loc*W-0.5
            const float x = (rx + ox[idx] / (float)W) * (float)W - 0.5f;
            const float y = (ry + oy[idx] / (float)H) * (float)H - 0.5f;
            const float xf = floorf(x), yf = floorf(y);
            const int   x0 = (int)xf, y0 = (int)yf;
            const float fx = x - xf, fy = y - yf;

            const bool xin0 = (x0 >= 0)     && (x0 <= W - 1);
            const bool xin1 = (x0 + 1 >= 0) && (x0 + 1 <= W - 1);
            const bool yin0 = (y0 >= 0)     && (y0 <= H - 1);
            const bool yin1 = (y0 + 1 >= 0) && (y0 + 1 <= H - 1);

            float v00 = 0.f, v01 = 0.f, v10 = 0.f, v11 = 0.f;
            if (yin0 && xin0) v00 = vbase[(size_t)(y0 * W + x0) * D_MODEL];
            if (yin0 && xin1) v01 = vbase[(size_t)(y0 * W + x0 + 1) * D_MODEL];
            if (yin1 && xin0) v10 = vbase[(size_t)((y0 + 1) * W + x0) * D_MODEL];
            if (yin1 && xin1) v11 = vbase[(size_t)((y0 + 1) * W + x0 + 1) * D_MODEL];

            const float s = (1.f - fy) * ((1.f - fx) * v00 + fx * v01)
                          +        fy  * ((1.f - fx) * v10 + fx * v11);
            acc = fmaf(l[idx] * inv, s, acc);
        }
    }

    out[bq * D_MODEL + h * D_HEAD + lane] = acc;
}

// ---------------------------------------------------------------------------
extern "C" void kernel_launch(void* const* d_in, const int* in_sizes, int n_in,
                              void* d_out, int out_size, void* d_ws, size_t ws_size,
                              hipStream_t stream) {
    const float* query = (const float*)d_in[0];   // (2, 13294, 256)
    const float* refpt = (const float*)d_in[1];   // (2, 13294, 4, 2)
    const float* value = (const float*)d_in[2];   // (2, 13294, 256)
    const float* W_off = (const float*)d_in[3];   // (256, 256)
    const float* b_off = (const float*)d_in[4];   // (256,)
    const float* W_att = (const float*)d_in[5];   // (256, 128)
    const float* b_att = (const float*)d_in[6];   // (128,)
    float*       out   = (float*)d_out;           // (2, 13294, 256)

    // ws layout: [ off : 26588*256 f32 ][ attn : 26588*128 f32 ][ packedB f16 ]
    float*    ws_off    = (float*)d_ws;
    float*    ws_attn   = ws_off + (size_t)M_ROWS * NC_OFF;
    _Float16* ws_packed = (_Float16*)(ws_attn + (size_t)M_ROWS * NC_ATTN);

    // Phase 0: one-time weight pack into WMMA B layout (196 KB, L2-resident)
    pack_b<<<N_TILES, 256, 0, stream>>>(W_off, W_att, ws_packed);

    // Phase 1: WMMA projections (offsets + attention logits)
    dim3 g1((M_TILES + WAVES_PB - 1) / WAVES_PB, N_TILES / NT_GROUP);  // (208, 4)
    proj_wmma<<<g1, 256, 0, stream>>>(query, ws_packed, b_off, b_att,
                                      ws_off, ws_attn);

    // Phase 2: softmax + deformable bilinear sampling.
    // One wave per (b, q, h): 2*13294*8 = 212704 waves, 8 waves per 256-thr block.
    const int total_waves = N_BATCH * LEN_Q * N_HEADS;   // 212704
    const int blocks      = total_waves / 8;             // 26588 exact
    msda_sample<<<blocks, 256, 0, stream>>>(value, refpt, ws_off, ws_attn, out);

    (void)in_sizes; (void)n_in; (void)out_size; (void)ws_size;
}
